// GAT_80058190398002
// MI455X (gfx1250) — matrix operand order repaired and checked by reading, not compiled, not run on hardware
//
#include <hip/hip_runtime.h>

typedef __bf16 bf16_t;
typedef __attribute__((ext_vector_type(16))) __bf16 v16bf;
typedef __attribute__((ext_vector_type(8)))  __bf16 v8bf;
typedef __attribute__((ext_vector_type(8)))  float  v8f;
typedef int gat_v4i __attribute__((vector_size(16)));

#define GAT_N    30000
#define GAT_E    480000
#define GAT_EALL (GAT_E + GAT_N)
#define GAT_C    128
#define GAT_H    3
#define GAT_M    384   // H*C
#define GAT_OUT  8

#define LDS_ROWS   192          // half of W rows staged per block
#define LDS_STRIDE 136          // 128 + 8 bf16 pad -> 272B rows, uniform LDS bank spread

#if defined(__AMDGCN__) && __has_builtin(__builtin_amdgcn_global_load_async_to_lds_b128)
#define GAT_ASYNC_LDS 1
#endif

// ---------------- float -> bf16 conversion (RTE via hardware cvt) ----------------
__global__ void f32_to_bf16_kernel(const float* __restrict__ in, bf16_t* __restrict__ out, int n) {
    int i = blockIdx.x * blockDim.x + threadIdx.x;
    if (i < n) out[i] = (bf16_t)in[i];
}

// 16-byte global -> LDS copy; async (ASYNCcnt) path when available
__device__ __forceinline__ void stage16(const bf16_t* g, bf16_t* l) {
#if GAT_ASYNC_LDS
    __builtin_amdgcn_global_load_async_to_lds_b128(
        (__attribute__((address_space(1))) gat_v4i*)g,
        (__attribute__((address_space(3))) gat_v4i*)l, 0, 0);
#else
    *(float4*)l = *(const float4*)g;
#endif
}

__device__ __forceinline__ void stage_wait() {
#if GAT_ASYNC_LDS
#if __has_builtin(__builtin_amdgcn_s_wait_asynccnt)
    __builtin_amdgcn_s_wait_asynccnt(0);
#else
    asm volatile("s_wait_asynccnt 0x0" ::: "memory");
#endif
#endif
}

// ---------------- WMMA GEMM: xs[N,384] = xb[N,128] @ wb[384,128]^T ----------------
// Block = 128 threads (4 waves). grid.y in {0,1} selects which 192 W rows are staged
// in LDS. Each wave keeps its A fragments in registers and sweeps 3 col tiles
// against LDS-resident B; blocks grid-stride over node row tiles.
__global__ __launch_bounds__(128) void gemm_xw_bf16_lds_kernel(
    const bf16_t* __restrict__ xb,   // [N,128] row-major
    const bf16_t* __restrict__ wb,   // [384,128] row-major
    float* __restrict__ xs,          // [N,384] row-major
    int nrow_tiles)
{
    __shared__ __align__(16) bf16_t wsh[LDS_ROWS * LDS_STRIDE];   // 51 KB

    const int tid  = threadIdx.x;
    const int lane = tid & 31;
    const int wave = tid >> 5;
    const int wrow0 = blockIdx.y * LDS_ROWS;

    // ---- stage 192x128 bf16 W rows into padded LDS (192*16 chunks of 16B) ----
    for (int idx = tid; idx < LDS_ROWS * 16; idx += 128) {
        int row = idx >> 4, ch = idx & 15;
        stage16(wb + (size_t)(wrow0 + row) * GAT_C + ch * 8,
                wsh + row * LDS_STRIDE + ch * 8);
    }
    stage_wait();
    __syncthreads();

    const int r    = lane & 15;
    const int half = lane >> 4;
    const int ct0  = wave * 3;     // local col tiles [3w, 3w+2] of the 12 in this half

    for (int rt = blockIdx.x; rt < nrow_tiles; rt += gridDim.x) {
        const int row_base = rt * 16;
        const bf16_t* arow = xb + (size_t)(row_base + r) * GAT_C;

        // A fragments (16x32 bf16 each): lanes 0-15 carry K {0..7,16..23},
        // lanes 16-31 K {8..15,24..31}; kept in registers for all 3 col tiles.
        v16bf afrag[4];
#pragma unroll
        for (int kb = 0; kb < 4; ++kb) {
            v8bf alo = *(const v8bf*)(arow + kb * 32 + half * 8);
            v8bf ahi = *(const v8bf*)(arow + kb * 32 + 16 + half * 8);
#pragma unroll
            for (int i = 0; i < 8; ++i) { afrag[kb][i] = alo[i]; afrag[kb][i + 8] = ahi[i]; }
        }

#pragma unroll
        for (int t = 0; t < 3; ++t) {
            const int cl = (ct0 + t) * 16;               // local col base in LDS
            const bf16_t* brow = wsh + (cl + r) * LDS_STRIDE;
            v8f c = {};
#pragma unroll
            for (int kb = 0; kb < 4; ++kb) {
                // B fragment (32x16): lane n holds column n; lanes 0-15 K {0..7,16..23}, lanes 16-31 K {8..15,24..31}
                v8bf blo = *(const v8bf*)(brow + kb * 32 + half * 8);
                v8bf bhi = *(const v8bf*)(brow + kb * 32 + 16 + half * 8);
                v16bf b = __builtin_shufflevector(blo, bhi, 0, 1, 2, 3, 4, 5, 6, 7, 8, 9, 10, 11, 12, 13, 14, 15);
                c = __builtin_amdgcn_wmma_f32_16x16x32_bf16(false, afrag[kb], false, b,
                                                            (short)0, c, false, false);
            }
            const int cg = wrow0 + cl;                   // global col base
#pragma unroll
            for (int j = 0; j < 8; ++j)
                xs[(size_t)(row_base + j + 8 * half) * GAT_M + cg + r] = c[j];
        }
    }
}

// ---------------- per-node attention logits: one wave per (node, head) ----------------
__global__ __launch_bounds__(256) void node_alpha_kernel(
    const float* __restrict__ xs, const float* __restrict__ att_src,
    const float* __restrict__ att_dst, float* __restrict__ as_out,
    float* __restrict__ ad_out, int n)
{
    int gw   = (blockIdx.x * 256 + threadIdx.x) >> 5;   // (node*H + h)
    int lane = threadIdx.x & 31;
    if (gw >= n * GAT_H) return;
    int node = gw / GAT_H, h = gw % GAT_H;
    float4 v = ((const float4*)(xs + (size_t)node * GAT_M + h * GAT_C))[lane];
    float4 a = ((const float4*)(att_src + h * GAT_C))[lane];
    float4 b = ((const float4*)(att_dst + h * GAT_C))[lane];
    float s = v.x * a.x + v.y * a.y + v.z * a.z + v.w * a.w;
    float d = v.x * b.x + v.y * b.y + v.z * b.z + v.w * b.w;
#pragma unroll
    for (int off = 16; off > 0; off >>= 1) {
        s += __shfl_down(s, off, 32);
        d += __shfl_down(d, off, 32);
    }
    if (lane == 0) { as_out[gw] = s; ad_out[gw] = d; }
}

// ke[h] = sum_c We[h*C+c]*att_e[h,c]  (edge-logit collapses to ea[e]*ke[h])
__global__ void compute_ke_kernel(const float* __restrict__ We, const float* __restrict__ att_e,
                                  float* __restrict__ ke)
{
    int h = threadIdx.x;
    if (h < GAT_H) {
        float s = 0.f;
        for (int c = 0; c < GAT_C; ++c) s += We[h * GAT_C + c] * att_e[h * GAT_C + c];
        ke[h] = s;
    }
}

// ---------------- edge_attr mean ----------------
__global__ void ea_sum_kernel(const float* __restrict__ ea, float* __restrict__ sum, int n) {
    __shared__ float sm[256];
    int i = blockIdx.x * blockDim.x + threadIdx.x;
    sm[threadIdx.x] = (i < n) ? ea[i] : 0.f;
    __syncthreads();
    for (int s = 128; s > 0; s >>= 1) {
        if ((int)threadIdx.x < s) sm[threadIdx.x] += sm[threadIdx.x + s];
        __syncthreads();
    }
    if (threadIdx.x == 0) atomicAdd(sum, sm[0]);
}

__global__ void init_md_kernel(float* __restrict__ m, float* __restrict__ denom, int n3) {
    int i = blockIdx.x * blockDim.x + threadIdx.x;
    if (i < n3) { m[i] = -1e30f; denom[i] = 0.f; }
}

__device__ __forceinline__ void atomicMaxFloat(float* addr, float val) {
    if (val >= 0.f)
        atomicMax((int*)addr, __float_as_int(val));
    else
        atomicMin((unsigned int*)addr, (unsigned int)__float_as_int(val));
}

__device__ __forceinline__ void edge_src_dst(const int* ei, int e, float ea_mean,
                                             const float* ea, int& s, int& d, float& eav) {
    if (e < GAT_E) { s = ei[e]; d = ei[GAT_E + e]; eav = ea[e]; }
    else           { s = d = e - GAT_E; eav = ea_mean; }
}

// ---------------- segment softmax: max pass ----------------
__global__ void edge_max_kernel(const int* __restrict__ ei, const float* __restrict__ ea,
                                const float* __restrict__ easum,
                                const float* __restrict__ as_in, const float* __restrict__ ad_in,
                                const float* __restrict__ ke, float* __restrict__ m)
{
    int e = blockIdx.x * blockDim.x + threadIdx.x;
    if (e >= GAT_EALL) return;
    int s, d; float eav;
    edge_src_dst(ei, e, easum[0] * (1.f / GAT_E), ea, s, d, eav);
#pragma unroll
    for (int h = 0; h < GAT_H; ++h) {
        float a = as_in[s * GAT_H + h] + ad_in[d * GAT_H + h] + eav * ke[h];
        a = a > 0.f ? a : 0.2f * a;          // leaky_relu(0.2)
        atomicMaxFloat(&m[d * GAT_H + h], a);
    }
}

// ---------------- segment softmax: exp + denom pass ----------------
__global__ void edge_exp_kernel(const int* __restrict__ ei, const float* __restrict__ ea,
                                const float* __restrict__ easum,
                                const float* __restrict__ as_in, const float* __restrict__ ad_in,
                                const float* __restrict__ ke, const float* __restrict__ m,
                                float* __restrict__ exbuf, float* __restrict__ denom)
{
    int e = blockIdx.x * blockDim.x + threadIdx.x;
    if (e >= GAT_EALL) return;
    int s, d; float eav;
    edge_src_dst(ei, e, easum[0] * (1.f / GAT_E), ea, s, d, eav);
#pragma unroll
    for (int h = 0; h < GAT_H; ++h) {
        float a = as_in[s * GAT_H + h] + ad_in[d * GAT_H + h] + eav * ke[h];
        a = a > 0.f ? a : 0.2f * a;
        float ex = __expf(a - m[d * GAT_H + h]);
        exbuf[e * GAT_H + h] = ex;
        atomicAdd(&denom[d * GAT_H + h], ex);
    }
}

// ---------------- weighted scatter: acc[dst,h,:] += att * xs[src,h,:] ----------------
// 8 waves/block, one wave per (edge, head) work item, grid-stride.
__global__ __launch_bounds__(256) void edge_acc_kernel(
    const int* __restrict__ ei, const float* __restrict__ exbuf, const float* __restrict__ denom,
    const float* __restrict__ xs, float* __restrict__ acc)
{
    int lane   = threadIdx.x & 31;
    int gwave  = (blockIdx.x * 256 + threadIdx.x) >> 5;
    int nwaves = gridDim.x * 8;
    for (int idx = gwave; idx < GAT_EALL * GAT_H; idx += nwaves) {
        int e = idx / GAT_H, h = idx - e * GAT_H;
        int s, d;
        if (e < GAT_E) { s = ei[e]; d = ei[GAT_E + e]; } else { s = d = e - GAT_E; }
        float att = exbuf[idx] / denom[d * GAT_H + h];
        const float4 v = ((const float4*)(xs + (size_t)s * GAT_M + h * GAT_C))[lane];
        float* dstp = acc + (size_t)d * GAT_M + h * GAT_C + lane * 4;
        atomicAdd(dstp + 0, att * v.x);
        atomicAdd(dstp + 1, att * v.y);
        atomicAdd(dstp + 2, att * v.z);
        atomicAdd(dstp + 3, att * v.w);
    }
}

// ---------------- head-mean + bias + leaky(0.25), emit fp32 + bf16 ----------------
__global__ void finalize_kernel(const float* __restrict__ acc, const float* __restrict__ bias,
                                float* __restrict__ hout, bf16_t* __restrict__ hb, int n)
{
    int idx = blockIdx.x * blockDim.x + threadIdx.x;
    if (idx >= n * GAT_C) return;
    int node = idx >> 7, c = idx & 127;
    const float* a = acc + (size_t)node * GAT_M;
    float v = (a[c] + a[GAT_C + c] + a[2 * GAT_C + c]) * (1.f / 3.f) + bias[c];
    v = v > 0.f ? v : 0.25f * v;
    hout[idx] = v;
    hb[idx] = (bf16_t)v;
}

// ---------------- classifier: out[N,8] = h @ Wc^T + bc ----------------
__global__ void classifier_kernel(const float* __restrict__ h, const float* __restrict__ Wc,
                                  const float* __restrict__ bc, float* __restrict__ out, int n)
{
    int idx = blockIdx.x * blockDim.x + threadIdx.x;  // n*OUT + o
    if (idx >= n * GAT_OUT) return;
    int node = idx >> 3, o = idx & 7;
    const float* row = h + (size_t)node * GAT_C;
    const float* w = Wc + o * GAT_C;
    float s = bc[o];
#pragma unroll 4
    for (int c = 0; c < GAT_C; ++c) s += row[c] * w[c];
    out[idx] = s;
}

extern "C" void kernel_launch(void* const* d_in, const int* in_sizes, int n_in,
                              void* d_out, int out_size, void* d_ws, size_t ws_size,
                              hipStream_t stream) {
    (void)in_sizes; (void)n_in; (void)out_size; (void)ws_size;
    const float* x  = (const float*)d_in[0];
    const int*   ei = (const int*)d_in[1];
    const float* ea = (const float*)d_in[2];
    const float *Wl[3], *asl[3], *adl[3], *Wel[3], *ael[3], *bl[3];
    for (int l = 0; l < 3; ++l) {
        int base = 3 + 6 * l;
        Wl[l]  = (const float*)d_in[base + 0];
        asl[l] = (const float*)d_in[base + 1];
        adl[l] = (const float*)d_in[base + 2];
        Wel[l] = (const float*)d_in[base + 3];
        ael[l] = (const float*)d_in[base + 4];
        bl[l]  = (const float*)d_in[base + 5];
    }
    const float* Wc = (const float*)d_in[21];
    const float* bc = (const float*)d_in[22];

    // workspace layout (256B aligned slices)
    char* ws = (char*)d_ws;
    size_t off = 0;
    auto carve = [&](size_t bytes) { size_t r = off; off = (off + bytes + 255) & ~(size_t)255; return r; };
    bf16_t* xb   = (bf16_t*)(ws + carve((size_t)GAT_N * GAT_C * 2));
    bf16_t* wb   = (bf16_t*)(ws + carve((size_t)GAT_M * GAT_C * 2));
    float*  xs   = (float*) (ws + carve((size_t)GAT_N * GAT_M * 4));
    float*  acc  = (float*) (ws + carve((size_t)GAT_N * GAT_M * 4));
    float*  asb  = (float*) (ws + carve((size_t)GAT_N * GAT_H * 4));
    float*  adb  = (float*) (ws + carve((size_t)GAT_N * GAT_H * 4));
    float*  mbuf = (float*) (ws + carve((size_t)GAT_N * GAT_H * 4));
    float*  dnm  = (float*) (ws + carve((size_t)GAT_N * GAT_H * 4));
    float*  exb  = (float*) (ws + carve((size_t)GAT_EALL * GAT_H * 4));
    float*  hbuf = (float*) (ws + carve((size_t)GAT_N * GAT_C * 4));
    float*  ke   = (float*) (ws + carve(64));
    float*  easm = (float*) (ws + carve(64));

    // edge_attr mean (sum once; kernels divide by E)
    (void)hipMemsetAsync(easm, 0, sizeof(float), stream);
    ea_sum_kernel<<<(GAT_E + 255) / 256, 256, 0, stream>>>(ea, easm, GAT_E);

    // layer-1 input features -> bf16
    f32_to_bf16_kernel<<<(GAT_N * GAT_C + 255) / 256, 256, 0, stream>>>(x, xb, GAT_N * GAT_C);

    for (int l = 0; l < 3; ++l) {
        f32_to_bf16_kernel<<<(GAT_M * GAT_C + 255) / 256, 256, 0, stream>>>(Wl[l], wb, GAT_M * GAT_C);
        // xs = xb @ wb^T via WMMA; grid.y selects which half of W is LDS-resident
        gemm_xw_bf16_lds_kernel<<<dim3(625, 2), 128, 0, stream>>>(xb, wb, xs, GAT_N / 16);

        node_alpha_kernel<<<(GAT_N * GAT_H * 32 + 255) / 256, 256, 0, stream>>>(xs, asl[l], adl[l], asb, adb, GAT_N);
        compute_ke_kernel<<<1, 32, 0, stream>>>(Wel[l], ael[l], ke);

        init_md_kernel<<<(GAT_N * GAT_H + 255) / 256, 256, 0, stream>>>(mbuf, dnm, GAT_N * GAT_H);
        (void)hipMemsetAsync(acc, 0, (size_t)GAT_N * GAT_M * 4, stream);

        edge_max_kernel<<<(GAT_EALL + 255) / 256, 256, 0, stream>>>(ei, ea, easm, asb, adb, ke, mbuf);
        edge_exp_kernel<<<(GAT_EALL + 255) / 256, 256, 0, stream>>>(ei, ea, easm, asb, adb, ke, mbuf, exb, dnm);
        edge_acc_kernel<<<4096, 256, 0, stream>>>(ei, exb, dnm, xs, acc);

        finalize_kernel<<<(GAT_N * GAT_C + 255) / 256, 256, 0, stream>>>(acc, bl[l], hbuf, xb, GAT_N);
    }

    classifier_kernel<<<(GAT_N * GAT_OUT + 255) / 256, 256, 0, stream>>>(hbuf, Wc, bc, (float*)d_out, GAT_N);
}
